// ResNodeConv_82978768159520
// MI455X (gfx1250) — matrix operand order repaired and checked by reading, not compiled, not run on hardware
//
#include <hip/hip_runtime.h>
#include <hip/hip_bf16.h>

typedef __attribute__((ext_vector_type(16))) __bf16 v16bf;
typedef __attribute__((ext_vector_type(8)))  float  v8f;

#define N_NODES 50000
#define N_EDGES 600000
#define C 128
#define EPSV 1e-8f
#define BN_EPS 1e-5f

union BFrag { v16bf v; unsigned int u[8]; uint4 q[2]; };

__device__ __forceinline__ unsigned int pack_bf16(float lo, float hi) {
  unsigned int a = __float_as_uint(lo);
  unsigned int b = __float_as_uint(hi);
  a += 0x7FFFu + ((a >> 16) & 1u);          // round-to-nearest-even
  b += 0x7FFFu + ((b >> 16) & 1u);
  return (a >> 16) | (b & 0xFFFF0000u);
}

// fast sigmoid: 1/(1+e^-x) via v_exp_f32 + v_rcp_f32
__device__ __forceinline__ float sigmoidf(float v) {
  return __builtin_amdgcn_rcpf(1.0f + __expf(-v));
}

// ---------------- zero workspace ----------------
__global__ void k_zero(float* __restrict__ p, int n4) {
  int i = blockIdx.x * blockDim.x + threadIdx.x;
  int stride = gridDim.x * blockDim.x;
  float4 z = make_float4(0.f, 0.f, 0.f, 0.f);
  for (; i < n4; i += stride) ((float4*)p)[i] = z;
}

// ---------------- f32 -> bf16 (packed pairs), one-time prep ----------------
__global__ void k_cvt(const float* __restrict__ src, unsigned int* __restrict__ dst,
                      int n2) {
  int i = blockIdx.x * blockDim.x + threadIdx.x;
  int stride = gridDim.x * blockDim.x;
  for (; i < n2; i += stride) {
    float2 f = ((const float2*)src)[i];
    dst[i] = pack_bf16(f.x, f.y);
  }
}

// ---------------- denom = segment_sum(sigmoid(edge_attr), tgt) ----------------
__global__ void k_denom(const float* __restrict__ ea, const int* __restrict__ eidx,
                        float* __restrict__ denom) {
  int i = blockIdx.x * blockDim.x + threadIdx.x;
  if (i >= N_EDGES * 32) return;
  int e  = i >> 5;
  int c4 = (i & 31) << 2;
  float4 v = *(const float4*)(ea + (size_t)e * C + c4);
  int tg = eidx[N_EDGES + e];
  float* d = denom + (size_t)tg * C + c4;
  atomicAdd(d + 0, sigmoidf(v.x));
  atomicAdd(d + 1, sigmoidf(v.y));
  atomicAdd(d + 2, sigmoidf(v.z));
  atomicAdd(d + 3, sigmoidf(v.w));
}

// Build A fragments straight from a bf16 row: contiguous K order,
// lane half 0 -> K = 0..15, half 1 -> K = 16..31 (per 32-K step).
// bf16 pairs are already packed little-endian => pure b128 loads, no VALU.
__device__ __forceinline__ void build_a_bf(const unsigned short* __restrict__ xr,
                                           int hlf, BFrag (&a)[4]) {
#pragma unroll
  for (int s = 0; s < 4; ++s) {
    const int kb = s * 32 + hlf * 16;                   // 16B-aligned
    a[s].q[0] = *(const uint4*)(xr + kb);               // K = kb..kb+7
    a[s].q[1] = *(const uint4*)(xr + kb + 8);           // K = kb+8..kb+15
  }
}

// B fragment from LDS W (bf16, row-major [n][k]): lane L holds column
// n = t*16 + (L&15), K = s*32 + (L>>4)*16 .. +15 (consecutive, pairs packed).
__device__ __forceinline__ v16bf load_b(const unsigned short* __restrict__ Wlds,
                                        int t, int s, int m16, int hlf) {
  BFrag b;
  const int n  = t * 16 + m16;
  const int kb = s * 32 + hlf * 16;
  const uint4* bp = (const uint4*)&Wlds[n * C + kb];   // 32B aligned
  b.q[0] = bp[0];
  b.q[1] = bp[1];
  return b.v;
}

// Copy pre-converted bf16 W (32KB) global -> LDS, no conversion math.
__device__ __forceinline__ void fill_wlds(unsigned short* Wlds,
                                          const unsigned short* __restrict__ Wb) {
  const uint4* g = (const uint4*)Wb;
  uint4* l = (uint4*)Wlds;
  for (int i = threadIdx.x; i < (C * C) / 8; i += 256) l[i] = g[i];
}

// ---------------- edge GEMM: agg += atten * (x[src] @ W_gea^T) ----------------
__global__ void __launch_bounds__(256) k_gea(
    const unsigned short* __restrict__ xb, const float* __restrict__ ea,
    const int* __restrict__ eidx, const unsigned short* __restrict__ Wb,
    const float* __restrict__ denom, float* __restrict__ agg) {
  __shared__ unsigned short Wlds[C * C];
  fill_wlds(Wlds, Wb);
  __syncthreads();

  const int wave = threadIdx.x >> 5;
  const int lane = threadIdx.x & 31;
  const int hlf  = lane >> 4;
  const int m16  = lane & 15;
  const int gbase = (blockIdx.x * 8 + wave) * 16;
  if (gbase >= N_EDGES) return;

  const int srcRow = eidx[gbase + m16];
  BFrag a[4];
  build_a_bf(xb + (size_t)srcRow * C, hlf, a);

  // This lane's accumulator rows are edges gbase + hlf*8 + r (r=0..7);
  // hoist their edge_attr / target-row offsets out of the n-tile loop.
  size_t eoff[8];
  size_t toff[8];
#pragma unroll
  for (int r = 0; r < 8; ++r) {
    const int e = gbase + hlf * 8 + r;
    eoff[r] = (size_t)e * C;
    toff[r] = (size_t)eidx[N_EDGES + e] * C;
  }

#pragma unroll
  for (int t = 0; t < 8; ++t) {
    v8f c = {};
#pragma unroll
    for (int s = 0; s < 4; ++s) {
      v16bf b = load_b(Wlds, t, s, m16, hlf);
      c = __builtin_amdgcn_wmma_f32_16x16x32_bf16(false, a[s].v, false, b,
                                                  (short)0, c, false, false);
    }
    const int ch = t * 16 + m16;  // C/D layout: lane holds column ch
#pragma unroll
    for (int r = 0; r < 8; ++r) {
      // atten = sigmoid(ea)/(dn+eps) = rcp((1+e^-ea)*(dn+eps))
      const float ev = 1.0f + __expf(-ea[eoff[r] + ch]);
      const float dn = denom[toff[r] + ch] + EPSV;
      const float atten = __builtin_amdgcn_rcpf(ev * dn);
      atomicAdd(agg + toff[r] + ch, c[r] * atten);
    }
  }
}

// ---------------- node GEMM: h = x @ W_lin^T + agg, + channel stats ----------------
__global__ void __launch_bounds__(256) k_lin(
    const unsigned short* __restrict__ xb, const unsigned short* __restrict__ Wb,
    const float* __restrict__ agg, float* __restrict__ h,
    float* __restrict__ gsum, float* __restrict__ gsq) {
  __shared__ unsigned short Wlds[C * C];
  __shared__ float ssum[C];
  __shared__ float ssq[C];
  fill_wlds(Wlds, Wb);
  if (threadIdx.x < C) { ssum[threadIdx.x] = 0.f; ssq[threadIdx.x] = 0.f; }
  __syncthreads();

  const int wave = threadIdx.x >> 5;
  const int lane = threadIdx.x & 31;
  const int hlf  = lane >> 4;
  const int m16  = lane & 15;
  const int rbase = (blockIdx.x * 8 + wave) * 16;
  if (rbase < N_NODES) {
    BFrag a[4];
    build_a_bf(xb + (size_t)(rbase + m16) * C, hlf, a);
#pragma unroll
    for (int t = 0; t < 8; ++t) {
      v8f c = {};
#pragma unroll
      for (int s = 0; s < 4; ++s) {
        v16bf b = load_b(Wlds, t, s, m16, hlf);
        c = __builtin_amdgcn_wmma_f32_16x16x32_bf16(false, a[s].v, false, b,
                                                    (short)0, c, false, false);
      }
      const int ch = t * 16 + m16;
      float lsum = 0.f, lsq = 0.f;
#pragma unroll
      for (int r = 0; r < 8; ++r) {
        const int row = rbase + hlf * 8 + r;
        const float hv = c[r] + agg[(size_t)row * C + ch];
        h[(size_t)row * C + ch] = hv;
        lsum += hv;
        lsq  += hv * hv;
      }
      atomicAdd(&ssum[ch], lsum);  // ds_add_f32
      atomicAdd(&ssq[ch],  lsq);
    }
  }
  __syncthreads();
  if (threadIdx.x < C) {
    atomicAdd(&gsum[threadIdx.x], ssum[threadIdx.x]);
    atomicAdd(&gsq[threadIdx.x],  ssq[threadIdx.x]);
  }
}

// ---------------- BN (batch stats) + ReLU + residual ----------------
__global__ void k_bn(const float* __restrict__ x, const float* __restrict__ h,
                     const float* __restrict__ gsum, const float* __restrict__ gsq,
                     const float* __restrict__ gamma, const float* __restrict__ beta,
                     float* __restrict__ out) {
  int i = blockIdx.x * blockDim.x + threadIdx.x;
  if (i >= N_NODES * 32) return;
  int row = i >> 5;
  int c4  = (i & 31) << 2;
  const float invN = 1.0f / (float)N_NODES;
  float4 hv = *(const float4*)(h + (size_t)row * C + c4);
  float4 xv = *(const float4*)(x + (size_t)row * C + c4);
  float hh[4] = {hv.x, hv.y, hv.z, hv.w};
  float xx[4] = {xv.x, xv.y, xv.z, xv.w};
  float oo[4];
#pragma unroll
  for (int j = 0; j < 4; ++j) {
    int ch = c4 + j;
    float mean = gsum[ch] * invN;
    float var  = gsq[ch] * invN - mean * mean;  // biased, matches torch BN
    float inv  = __builtin_amdgcn_rsqf(var + BN_EPS);
    float v    = (hh[j] - mean) * inv * gamma[ch] + beta[ch];
    oo[j] = xx[j] + fmaxf(v, 0.0f);
  }
  *(float4*)(out + (size_t)row * C + c4) = make_float4(oo[0], oo[1], oo[2], oo[3]);
}

extern "C" void kernel_launch(void* const* d_in, const int* in_sizes, int n_in,
                              void* d_out, int out_size, void* d_ws, size_t ws_size,
                              hipStream_t stream) {
  const float* x     = (const float*)d_in[0];
  const float* eattr = (const float*)d_in[1];
  const int*   eidx  = (const int*)d_in[2];
  const float* Wgea  = (const float*)d_in[3];
  const float* Wlin  = (const float*)d_in[4];
  const float* gamma = (const float*)d_in[5];
  const float* beta  = (const float*)d_in[6];
  float* out = (float*)d_out;

  const size_t NC = (size_t)N_NODES * C;
  float* wsf   = (float*)d_ws;
  float* denom = wsf;                       // [NC]
  float* agg   = wsf + NC;                  // [NC]
  float* h     = wsf + 2 * NC;              // [NC]
  float* gsum  = wsf + 3 * NC;              // [C]
  float* gsq   = wsf + 3 * NC + C;          // [C]
  // bf16 staging (16B-aligned offsets)
  unsigned short* xb  = (unsigned short*)(wsf + 3 * NC + 2 * C);  // [NC] bf16
  unsigned short* Wgb = xb + NC;                                  // [C*C] bf16
  unsigned short* Wlb = Wgb + C * C;                              // [C*C] bf16

  // zero denom/agg/stats
  int n4 = (int)((2 * NC + 2 * C) / 4);
  k_zero<<<2048, 256, 0, stream>>>(wsf, n4);
  k_zero<<<1, 64, 0, stream>>>(gsum, (2 * C) / 4);

  // one-time bf16 conversions
  k_cvt<<<1024, 256, 0, stream>>>(x, (unsigned int*)xb, (int)(NC / 2));
  k_cvt<<<32, 256, 0, stream>>>(Wgea, (unsigned int*)Wgb, (C * C) / 2);
  k_cvt<<<32, 256, 0, stream>>>(Wlin, (unsigned int*)Wlb, (C * C) / 2);

  k_denom<<<(N_EDGES * 32 + 255) / 256, 256, 0, stream>>>(eattr, eidx, denom);

  int geaBlocks = (N_EDGES / 16 + 7) / 8;   // 4688
  k_gea<<<geaBlocks, 256, 0, stream>>>(xb, eattr, eidx, Wgb, denom, agg);

  int linBlocks = (N_NODES / 16 + 7) / 8;   // 391
  k_lin<<<linBlocks, 256, 0, stream>>>(xb, Wlb, agg, h, gsum, gsq);

  k_bn<<<(N_NODES * 32 + 255) / 256, 256, 0, stream>>>(x, h, gsum, gsq,
                                                       gamma, beta, out);
}
